// ContinuousGenHyperConnections_90443421319350
// MI455X (gfx1250) — matrix operand order repaired and checked by reading, not compiled, not run on hardware
//
#include <hip/hip_runtime.h>

// ---------------- types ----------------
typedef __attribute__((ext_vector_type(16))) __bf16 v16bf;
typedef __attribute__((ext_vector_type(8)))  float  v8f;

#define DIM   2048
#define TOKW  16          // tokens per workgroup (WMMA M)
#define ROWP  2056        // padded bf16 row (stagger LDS banks, keep 16B align)
#define NTILE 3           // 48 features = 3 x 16
#define KITER 64          // 2048 / 32

// pack two floats to packed bf16 (lo = a, hi = b); round-half-away via +0x8000,
// then v_perm_b32 grabs the two high halves in one instruction.
__device__ __forceinline__ unsigned int pack2bf(float a, float b) {
    unsigned int ua = __float_as_uint(a) + 0x8000u;
    unsigned int ub = __float_as_uint(b) + 0x8000u;
#if __has_builtin(__builtin_amdgcn_perm)
    return __builtin_amdgcn_perm(ub, ua, 0x07060302u);  // bytes: b3 b2 a3 a2
#else
    return (ub & 0xffff0000u) | (ua >> 16);
#endif
}

__device__ __forceinline__ float4 f4_fma(float a, float4 b, float4 c) {
    return make_float4(fmaf(a, b.x, c.x), fmaf(a, b.y, c.y),
                       fmaf(a, b.z, c.z), fmaf(a, b.w, c.w));
}

// ---------------- weight pre-pack: bf16, WMMA-B fragment order ----------------
// B-frag order used here (contiguous k, same as A): lane<16 -> col n=lane, K 0..15 of the 32-run;
// lane>=16 -> K 16..31.  chunk c (8 bf16 = 16B): c=0 first 8, c=1 next 8.
// flat idx = ((tile*64 + kkg)*2 + c)*32 + lane ; 16 bytes each.
__global__ void pack_weights_kernel(const float* __restrict__ Wc,
                                    const float* __restrict__ Wd,
                                    const float* __restrict__ Wr,
                                    const float* __restrict__ Ww,
                                    const float* __restrict__ Wdtc,
                                    const float* __restrict__ Wdtd,
                                    uint4* __restrict__ wp) {
    int id = blockIdx.x * blockDim.x + threadIdx.x;
    if (id >= NTILE * KITER * 2 * 32) return;
    int l    = id & 31;
    int c    = (id >> 5) & 1;
    int kkg  = (id >> 6) & 63;
    int tile = id >> 12;
    int feat = tile * 16 + (l & 15);
    int k    = kkg * 32 + ((l >> 4) << 4) + c * 8;
    const float* src = nullptr;
    if      (feat < 16) src = Wc   + (size_t)feat * DIM;
    else if (feat < 32) src = Wd   + (size_t)(feat - 16) * DIM;
    else if (feat < 36) src = Wr   + (size_t)(feat - 32) * DIM;
    else if (feat < 40) src = Ww   + (size_t)(feat - 36) * DIM;
    else if (feat == 40) src = Wdtc;
    else if (feat == 41) src = Wdtd;
    float v[8];
    #pragma unroll
    for (int i = 0; i < 8; ++i) v[i] = src ? src[k + i] : 0.0f;
    uint4 pk;
    pk.x = pack2bf(v[0], v[1]);
    pk.y = pack2bf(v[2], v[3]);
    pk.z = pack2bf(v[4], v[5]);
    pk.w = pack2bf(v[6], v[7]);
    wp[id] = pk;
}

// ---------------- fused main kernel: 16 tokens per workgroup ----------------
__global__ __launch_bounds__(256) void hyperconn_fused_kernel(
    const float* __restrict__ x, const uint4* __restrict__ Wp,
    const float* __restrict__ read_in, const float* __restrict__ alpha_read_in,
    const float* __restrict__ write_out, const float* __restrict__ alpha_write_out,
    const float* __restrict__ conserv_A, const float* __restrict__ b_conv,
    const float* __restrict__ diss_A,    const float* __restrict__ b_diss,
    const float* __restrict__ log_dt_c,  const float* __restrict__ log_dt_d,
    const float* __restrict__ b_dt_c,    const float* __restrict__ b_dt_d,
    float* __restrict__ out)
{
    __shared__ __align__(16) unsigned short xn[TOKW][ROWP];   // bf16 normalized x
    __shared__ float Cpart[6][16][16];                        // per-wave partial C tiles
    __shared__ float A4s[TOKW][16];
    __shared__ float wrs[TOKW][4];
    __shared__ float wws[TOKW][4];

    const int tid = threadIdx.x;
    const int tok = tid >> 4;          // 0..15
    const int ln  = tid & 15;          // 0..15
    const size_t base = ((size_t)blockIdx.x * TOKW + tok) * DIM;
    const float4* xr4 = (const float4*)(x + base);

    // ---- phase 1: RMS norm -> packed bf16 into LDS ----
    float ss = 0.f;
    #pragma unroll 4
    for (int i = 0; i < 32; ++i) {
        float4 v = xr4[i * 16 + ln];
        ss += v.x * v.x + v.y * v.y + v.z * v.z + v.w * v.w;
    }
    ss += __shfl_xor(ss, 1); ss += __shfl_xor(ss, 2);
    ss += __shfl_xor(ss, 4); ss += __shfl_xor(ss, 8);
    const float rinv = rsqrtf(ss * (1.0f / (float)DIM) + 1.1920929e-7f);
    #pragma unroll 4
    for (int i = 0; i < 32; ++i) {
        float4 v = xr4[i * 16 + ln];
        int k = (i * 16 + ln) * 4;
        uint2 p;
        p.x = pack2bf(v.x * rinv, v.y * rinv);
        p.y = pack2bf(v.z * rinv, v.w * rinv);
        *(uint2*)&xn[tok][k] = p;
    }
    __syncthreads();

    // ---- phase 2: WMMA projection C[16 tok x 48 feat] = xn @ W_all^T ----
    const int wave = tid >> 5, lane = tid & 31;
    if (wave < 6) {
        const int tile = wave >> 1, kh = wave & 1;
        const int m  = lane & 15;
        const int hi = lane >> 4;
        v8f acc = {0.f, 0.f, 0.f, 0.f, 0.f, 0.f, 0.f, 0.f};
        #pragma unroll 4
        for (int kk = 0; kk < 32; ++kk) {
            const int kkg = kh * 32 + kk;
            const int k0  = kkg * 32;
            const int kA  = k0 + (hi << 4);           // A layout: lane<16 K0..15, lane>=16 K16..31
            union { uint4 q[2]; v16bf v; } au, bu;
            au.q[0] = *(const uint4*)&xn[m][kA];
            au.q[1] = *(const uint4*)&xn[m][kA + 8];
            const int bidx = ((tile * 64 + kkg) * 2) * 32 + lane;
            bu.q[0] = Wp[bidx];
            bu.q[1] = Wp[bidx + 32];
            acc = __builtin_amdgcn_wmma_f32_16x16x32_bf16(
                      false, au.v, false, bu.v, (short)0, acc, false, false);
        }
        #pragma unroll
        for (int r = 0; r < 8; ++r)
            Cpart[wave][r + (hi << 3)][m] = acc[r];
    }
    __syncthreads();

    // ---- phase 3: per-token 4x4 generator algebra (one thread per token) ----
    if (tid < TOKW) {
        const int t = tid;
        float cv[16], dv[16], Mm[16], Rm[16];
        #pragma unroll
        for (int f = 0; f < 16; ++f) {
            cv[f] = Cpart[0][t][f] + Cpart[1][t][f];
            dv[f] = Cpart[2][t][f] + Cpart[3][t][f];
            Mm[f] = conserv_A[f] + cv[f] + b_conv[f];
            Rm[f] = diss_A[f]    + dv[f] + b_diss[f];
        }
        float rd[4], wv[4];
        #pragma unroll
        for (int n = 0; n < 4; ++n) {
            rd[n] = Cpart[4][t][n]     + Cpart[5][t][n];
            wv[n] = Cpart[4][t][4 + n] + Cpart[5][t][4 + n];
        }
        const float cdt = Cpart[4][t][8] + Cpart[5][t][8];
        const float ddt = Cpart[4][t][9] + Cpart[5][t][9];
        const float zc  = log_dt_c[0] + cdt + b_dt_c[0];
        const float dtc = 1e-3f + 0.999f / (1.f + __expf(-zc));
        const float zd  = log_dt_d[0] + ddt + b_dt_d[0];
        const float dtd = 1e-3f + 0.999f / (1.f + __expf(-zd));
        #pragma unroll
        for (int i = 0; i < 4; ++i)
            #pragma unroll
            for (int j = 0; j < 4; ++j) {
                float a = dtc * (Mm[i * 4 + j] - Mm[j * 4 + i]);
                float k = 0.f;
                #pragma unroll
                for (int jj = 0; jj < 4; ++jj) k += Rm[i * 4 + jj] * Rm[j * 4 + jj];
                A4s[t][i * 4 + j] = a - dtd * 0.5f * k;   // / sqrt(N=4)
            }
        const float ar = alpha_read_in[0], aw = alpha_write_out[0];
        #pragma unroll
        for (int n = 0; n < 4; ++n) {
            wrs[t][n] = 1.f / (1.f + __expf(-(read_in[n] + ar * rd[n])));
            wws[t][n] = write_out[n] + aw * wv[n];
        }
    }
    __syncthreads();

    // ---- phase 4: stream mix hm=(I+A)h, read gate, write mix ----
    float A4[16], wr[4], ww[4];
    #pragma unroll
    for (int f = 0; f < 16; ++f) A4[f] = A4s[tok][f];
    #pragma unroll
    for (int n = 0; n < 4; ++n) { wr[n] = wrs[tok][n]; ww[n] = wws[tok][n]; }

    float4* outp = (float4*)(out + base);
    #pragma unroll 2
    for (int i = 0; i < 8; ++i) {
        const int d4 = i * 16 + ln;               // float4 index in a 512 block
        float4 h[4], hm[4];
        #pragma unroll
        for (int n = 0; n < 4; ++n) h[n] = xr4[n * 128 + d4];
        #pragma unroll
        for (int n = 0; n < 4; ++n) {
            hm[n] = h[n];
            #pragma unroll
            for (int j = 0; j < 4; ++j) hm[n] = f4_fma(A4[n * 4 + j], h[j], hm[n]);
        }
        float4 br = make_float4(0.f, 0.f, 0.f, 0.f);
        #pragma unroll
        for (int n = 0; n < 4; ++n) br = f4_fma(wr[n], hm[n], br);
        #pragma unroll
        for (int n = 0; n < 4; ++n) outp[n * 128 + d4] = f4_fma(ww[n], br, hm[n]);
    }
}

// ---------------- launch ----------------
extern "C" void kernel_launch(void* const* d_in, const int* in_sizes, int n_in,
                              void* d_out, int out_size, void* d_ws, size_t ws_size,
                              hipStream_t stream) {
    (void)in_sizes; (void)n_in; (void)out_size; (void)ws_size;
    const float* x               = (const float*)d_in[0];
    const float* read_in         = (const float*)d_in[1];
    const float* alpha_read_in   = (const float*)d_in[2];
    const float* write_out       = (const float*)d_in[3];
    const float* alpha_write_out = (const float*)d_in[4];
    const float* W_read          = (const float*)d_in[5];
    const float* W_write         = (const float*)d_in[6];
    const float* conserv_A       = (const float*)d_in[7];
    const float* W_conv          = (const float*)d_in[8];
    const float* b_conv          = (const float*)d_in[9];
    const float* diss_A          = (const float*)d_in[10];
    const float* W_diss          = (const float*)d_in[11];
    const float* b_diss          = (const float*)d_in[12];
    const float* log_dt_c        = (const float*)d_in[13];
    const float* log_dt_d        = (const float*)d_in[14];
    const float* W_dt_c          = (const float*)d_in[15];
    const float* b_dt_c          = (const float*)d_in[16];
    const float* W_dt_d          = (const float*)d_in[17];
    const float* b_dt_d          = (const float*)d_in[18];
    float* out = (float*)d_out;
    uint4* wp  = (uint4*)d_ws;   // 3*64*2*32*16 B = 192 KB

    const int packN = NTILE * KITER * 2 * 32;
    pack_weights_kernel<<<(packN + 255) / 256, 256, 0, stream>>>(
        W_conv, W_diss, W_read, W_write, W_dt_c, W_dt_d, wp);

    const int numBlocks = (4 * 4096) / TOKW;   // Bt / 16 = 1024
    hyperconn_fused_kernel<<<numBlocks, 256, 0, stream>>>(
        x, wp, read_in, alpha_read_in, write_out, alpha_write_out,
        conserv_A, b_conv, diss_A, b_diss,
        log_dt_c, log_dt_d, b_dt_c, b_dt_d, out);
}
